// GraphAttention_15960098472479
// MI455X (gfx1250) — matrix operand order repaired and checked by reading, not compiled, not run on hardware
//
#include <hip/hip_runtime.h>
#include <hip/hip_bf16.h>

#define NN 30000
#define EE 480000
#define GG 256
#define G3 768
#define SCALE_F 0.0625f   // 256^-0.5
#define LN_EPS 1e-5f

typedef __attribute__((ext_vector_type(4)))  float    v4f;
typedef __attribute__((ext_vector_type(8)))  float    v8f;
typedef __attribute__((ext_vector_type(8)))  _Float16 v8h;
typedef __attribute__((ext_vector_type(16))) _Float16 v16h;

// ---------------- workspace layout (bytes) ----------------
// xh   : [NN][GG]  f16   @ 0            (15,360,000 B)
// Wh   : [G3][GG]  f16   @ 15,360,000   (   393,216 B)  (transposed Wqkv)
// qkv  : [NN][G3]  f32   @ 15,753,216   (92,160,000 B)
// maxb : [NN][GG]  f32   @ 107,913,216  (30,720,000 B)
// zb   : [NN][GG]  f32   @ 138,633,216  (30,720,000 B)
#define OFF_XH   ((size_t)0)
#define OFF_WH   ((size_t)15360000)
#define OFF_QKV  ((size_t)15753216)
#define OFF_MAX  ((size_t)107913216)
#define OFF_Z    ((size_t)138633216)

// ---------------- 1) LayerNorm -> f16 (wave per node) ----------------
__global__ void gat_ln_f16(const float* __restrict__ s,
                           const float* __restrict__ gamma,
                           const float* __restrict__ beta,
                           _Float16* __restrict__ xh) {
  int wave = (blockIdx.x * blockDim.x + threadIdx.x) >> 5;
  int lane = threadIdx.x & 31;
  if (wave >= NN) return;
  const float* row = s + (size_t)wave * GG + lane * 8;
  // s is read exactly once: nontemporal load, keep L2 for qkv re-reads
  v4f a = __builtin_nontemporal_load((const v4f*)row);
  v4f b = __builtin_nontemporal_load((const v4f*)(row + 4));
  float sum = 0.f, sq = 0.f;
#pragma unroll
  for (int i = 0; i < 4; ++i) { sum += a[i] + b[i]; sq += a[i]*a[i] + b[i]*b[i]; }
#pragma unroll
  for (int m = 16; m > 0; m >>= 1) { sum += __shfl_xor(sum, m); sq += __shfl_xor(sq, m); }
  float mu   = sum * (1.0f / GG);
  float var  = sq * (1.0f / GG) - mu * mu;
  float rstd = rsqrtf(var + LN_EPS);
  v4f g0 = *(const v4f*)(gamma + lane * 8);
  v4f g1 = *(const v4f*)(gamma + lane * 8 + 4);
  v4f b0 = *(const v4f*)(beta + lane * 8);
  v4f b1 = *(const v4f*)(beta + lane * 8 + 4);
  v8h o;
#pragma unroll
  for (int i = 0; i < 4; ++i) {
    o[i]     = (_Float16)((a[i] - mu) * rstd * g0[i] + b0[i]);
    o[i + 4] = (_Float16)((b[i] - mu) * rstd * g1[i] + b1[i]);
  }
  *(v8h*)(xh + (size_t)wave * GG + lane * 8) = o;
}

// ---------------- 2) Wqkv [GG][G3] f32 -> Wh [G3][GG] f16 ----------------
__global__ void gat_wconv(const float* __restrict__ w, _Float16* __restrict__ wh) {
  int idx = blockIdx.x * blockDim.x + threadIdx.x;
  if (idx >= GG * G3) return;
  int k = idx / G3;
  int n = idx % G3;
  wh[(size_t)n * GG + k] = (_Float16)w[idx];
}

// ---------------- 3) qkv = xh @ Wh^T via v_wmma_f32_16x16x32_f16 ----------------
// one wave computes a 16-row x 64-col output tile (4 N-subtiles, A reused)
__global__ void gat_qkv_wmma(const _Float16* __restrict__ xh,
                             const _Float16* __restrict__ wh,
                             float* __restrict__ qkv) {
  const int ROW_TILES  = NN / 16;   // 1875
  const int COL_GROUPS = G3 / 64;   // 12
  int wave = (blockIdx.x * blockDim.x + threadIdx.x) >> 5;
  if (wave >= ROW_TILES * COL_GROUPS) return;
  int rowTile  = wave / COL_GROUPS;
  int colGroup = wave % COL_GROUPS;
  int lane = threadIdx.x & 31;
  int lh   = lane & 15;
  int khi  = lane >> 4;

  const _Float16* arow = xh + (size_t)(rowTile * 16 + lh) * GG;
  v8f acc[4];
#pragma unroll
  for (int t = 0; t < 4; ++t) acc[t] = (v8f){};

  for (int k0 = 0; k0 < GG; k0 += 32) {
    // A 16x32 f16: lanes 0-15 row M=lh K=0-15 (elems 0-15);
    //              lanes 16-31 same rows, K=16-31
    v8h alo = *(const v8h*)(arow + k0 + khi * 16);
    v8h ahi = *(const v8h*)(arow + k0 + khi * 16 + 8);
    v16h A;
#pragma unroll
    for (int i = 0; i < 8; ++i) { A[i] = alo[i]; A[i + 8] = ahi[i]; }
#pragma unroll
    for (int t = 0; t < 4; ++t) {
      // B 32x16 f16: lane lh holds column n, 16 contiguous K per lane half
      int n = colGroup * 64 + t * 16 + lh;
      const _Float16* bcol = wh + (size_t)n * GG + k0 + khi * 16;
      v8h blo = *(const v8h*)(bcol);
      v8h bhi = *(const v8h*)(bcol + 8);
      v16h B;
#pragma unroll
      for (int i = 0; i < 8; ++i) { B[i] = blo[i]; B[i + 8] = bhi[i]; }
      acc[t] = __builtin_amdgcn_wmma_f32_16x16x32_f16(
          false, A, false, B, (short)0, acc[t], false, false);
    }
  }
  // C/D layout: VGPR r -> lanes 0-15: (M=r, N=lh); lanes 16-31: (M=8+r, N=lh)
#pragma unroll
  for (int t = 0; t < 4; ++t) {
    int ocol = colGroup * 64 + t * 16 + lh;
#pragma unroll
    for (int r = 0; r < 8; ++r) {
      int orow = rowTile * 16 + khi * 8 + r;
      qkv[(size_t)orow * G3 + ocol] = acc[t][r];
    }
  }
}

// ---------------- 4) init max/-inf, z/0, out/0 ----------------
__global__ void gat_init(float* __restrict__ maxb, float* __restrict__ zb,
                         float* __restrict__ out) {
  int i = blockIdx.x * blockDim.x + threadIdx.x;
  if (i >= NN * GG) return;
  maxb[i] = -3.402823466e38f;
  zb[i] = 0.f;
  out[i] = 0.f;
}

// ---------------- 5) edge pass 1: segment max (wave per edge) ----------------
// unsafeAtomicMax lowers to the native GLOBAL_ATOMIC_MAX_NUM_F32 (no CAS loop)
__global__ void gat_edge_max(const float* __restrict__ qkv,
                             const int* __restrict__ src,
                             const int* __restrict__ dst,
                             float* __restrict__ maxb) {
  int e = (blockIdx.x * blockDim.x + threadIdx.x) >> 5;
  if (e >= EE) return;
  int lane = threadIdx.x & 31;
  int sN = src[e], dN = dst[e];
  const float* qrow = qkv + (size_t)sN * G3 + lane * 8;
  const float* krow = qkv + (size_t)dN * G3 + GG + lane * 8;
  float* mrow = maxb + (size_t)dN * GG + lane * 8;
  v4f q0 = *(const v4f*)qrow,       q1 = *(const v4f*)(qrow + 4);
  v4f k0 = *(const v4f*)krow,       k1 = *(const v4f*)(krow + 4);
#pragma unroll
  for (int i = 0; i < 4; ++i) {
    unsafeAtomicMax(mrow + i,     q0[i] * k0[i] * SCALE_F);
    unsafeAtomicMax(mrow + 4 + i, q1[i] * k1[i] * SCALE_F);
  }
}

// ---------------- 6) edge pass 2: exp, z += w, out += v*w ----------------
__global__ void gat_edge_acc(const float* __restrict__ qkv,
                             const int* __restrict__ src,
                             const int* __restrict__ dst,
                             const float* __restrict__ maxb,
                             float* __restrict__ zb,
                             float* __restrict__ out) {
  int e = (blockIdx.x * blockDim.x + threadIdx.x) >> 5;
  if (e >= EE) return;
  int lane = threadIdx.x & 31;
  int sN = src[e], dN = dst[e];
  const float* qrow = qkv + (size_t)sN * G3 + lane * 8;
  const float* krow = qkv + (size_t)dN * G3 + GG + lane * 8;
  const float* vrow = qkv + (size_t)sN * G3 + 2 * GG + lane * 8;
  const float* mrow = maxb + (size_t)dN * GG + lane * 8;
  float* zrow = zb  + (size_t)dN * GG + lane * 8;
  float* orow = out + (size_t)dN * GG + lane * 8;
  v4f q0 = *(const v4f*)qrow, q1 = *(const v4f*)(qrow + 4);
  v4f k0 = *(const v4f*)krow, k1 = *(const v4f*)(krow + 4);
  v4f v0 = *(const v4f*)vrow, v1 = *(const v4f*)(vrow + 4);
  v4f m0 = *(const v4f*)mrow, m1 = *(const v4f*)(mrow + 4);
#pragma unroll
  for (int i = 0; i < 4; ++i) {
    float w0 = __expf(q0[i] * k0[i] * SCALE_F - m0[i]);
    float w1 = __expf(q1[i] * k1[i] * SCALE_F - m1[i]);
    unsafeAtomicAdd(zrow + i,     w0);
    unsafeAtomicAdd(zrow + 4 + i, w1);
    unsafeAtomicAdd(orow + i,     v0[i] * w0);
    unsafeAtomicAdd(orow + 4 + i, v1[i] * w1);
  }
}

// ---------------- 7) normalize ----------------
__global__ void gat_final(const float* __restrict__ zb, float* __restrict__ out) {
  int i = blockIdx.x * blockDim.x + threadIdx.x;
  if (i >= NN * GG) return;
  float z = zb[i];
  out[i] = (z > 0.f) ? out[i] / z : 0.f;
}

extern "C" void kernel_launch(void* const* d_in, const int* in_sizes, int n_in,
                              void* d_out, int out_size, void* d_ws, size_t ws_size,
                              hipStream_t stream) {
  const float* s     = (const float*)d_in[0];
  const float* Wqkv  = (const float*)d_in[1];
  const float* gamma = (const float*)d_in[2];
  const float* beta  = (const float*)d_in[3];
  const int*   src   = (const int*)d_in[4];
  const int*   dst   = (const int*)d_in[5];
  float* out = (float*)d_out;

  char* ws = (char*)d_ws;
  _Float16* xh  = (_Float16*)(ws + OFF_XH);
  _Float16* wh  = (_Float16*)(ws + OFF_WH);
  float*    qkv = (float*)(ws + OFF_QKV);
  float*    mx  = (float*)(ws + OFF_MAX);
  float*    zb  = (float*)(ws + OFF_Z);

  // 1) LayerNorm + f16 cast: 30000 waves
  gat_ln_f16<<<(NN * 32 + 255) / 256, 256, 0, stream>>>(s, gamma, beta, xh);
  // 2) weight transpose/cast
  gat_wconv<<<(GG * G3 + 255) / 256, 256, 0, stream>>>(Wqkv, wh);
  // 3) WMMA GEMM: 1875*12 = 22500 waves
  gat_qkv_wmma<<<(22500 * 32 + 255) / 256, 256, 0, stream>>>(xh, wh, qkv);
  // 4) init reduction buffers + output
  gat_init<<<(NN * GG + 255) / 256, 256, 0, stream>>>(mx, zb, out);
  // 5) segment max over edges
  gat_edge_max<<<(EE * 32 + 255) / 256, 256, 0, stream>>>(qkv, src, dst, mx);
  // 6) exp + denominator + weighted numerator
  gat_edge_acc<<<(EE * 32 + 255) / 256, 256, 0, stream>>>(qkv, src, dst, mx, zb, out);
  // 7) divide
  gat_final<<<(NN * GG + 255) / 256, 256, 0, stream>>>(zb, out);
}
